// Policy_25477746000273
// MI455X (gfx1250) — matrix operand order repaired and checked by reading, not compiled, not run on hardware
//
#include <hip/hip_runtime.h>
#include <hip/hip_bf16.h>

// Graph-transformer edge scorer for MI455X (gfx1250, wave32, WMMA).
// All GEMM-shaped math runs through v_wmma_f32_16x16x32_f16.
// Flash-attention (no 4096x4096 score materialization), fused edge MLP.
// GEMM waves are register-blocked 16x64 (1 A-frag reused by 4 B-frags/K-step).
// Workspace requirement: ~13.8 MB.

#define NNODES 4096
#define NEDGES 131072
#define INDIM  10
#define HDIM   128
#define NLAYER 4
#define NHEADS 4
#define HEADD  32
#define FFDIM  512

typedef _Float16 v16h __attribute__((ext_vector_type(16)));
typedef _Float16 v8h  __attribute__((ext_vector_type(8)));
typedef float    v8f  __attribute__((ext_vector_type(8)));

__device__ __forceinline__ v8f vzero8() {
  v8f z;
#pragma unroll
  for (int i = 0; i < 8; ++i) z[i] = 0.f;
  return z;
}

__device__ __forceinline__ v8f wmma_f16(v16h a, v16h b, v8f c) {
  // D = A(16x32 f16) * B(32x16 f16) + C(16x16 f32)
  return __builtin_amdgcn_wmma_f32_16x16x32_f16(false, a, false, b, (short)0, c,
                                                false, false);
}

// A-fragment (16x32 f16), A row-major with leading dim lda.
// Lane l holds row m0+(l&15); K slots: [hi*8 .. hi*8+7] and [16+hi*8 .. 16+hi*8+7].
__device__ __forceinline__ v16h load_a_frag(const _Float16* __restrict__ A, int lda,
                                            int m0, int k0, int lane) {
  int m  = m0 + (lane & 15);
  int hi = (lane >> 4) << 3;  // 0 or 8
  const _Float16* p = A + (size_t)m * lda + k0 + hi;
  v8h a0 = *(const v8h*)p;
  v8h a1 = *(const v8h*)(p + 16);
  v16h r;
#pragma unroll
  for (int i = 0; i < 8; ++i) { r[i] = a0[i]; r[8 + i] = a1[i]; }
  return r;
}

// B-fragment (32x16 f16) from a K-contiguous ("transposed") layout Bt[n][k], ld=ldb.
// Lane l: column n0+(l&15), K slots: [hi*8 .. hi*8+7] and [16+hi*8 .. 16+hi*8+7].
__device__ __forceinline__ v16h load_b_frag(const _Float16* __restrict__ Bt, int ldb,
                                            int n0, int k0, int lane) {
  const _Float16* p = Bt + (size_t)(n0 + (lane & 15)) * ldb + k0 + ((lane >> 4) << 3);
  v8h b0 = *(const v8h*)p;
  v8h b1 = *(const v8h*)(p + 16);
  v16h r;
#pragma unroll
  for (int i = 0; i < 8; ++i) { r[i] = b0[i]; r[8 + i] = b1[i]; }
  return r;
}

// ---------------- weight prep ----------------
__global__ void k_transpose_f16(const float* __restrict__ W, _Float16* __restrict__ Wt,
                                int K, int Nc) {
  int idx = blockIdx.x * blockDim.x + threadIdx.x;
  if (idx >= K * Nc) return;
  int n = idx / K, k = idx - n * K;
  Wt[idx] = (_Float16)W[(size_t)k * Nc + n];
}

__global__ void k_f32_to_f16(const float* __restrict__ in, _Float16* __restrict__ out, int n) {
  int i = blockIdx.x * blockDim.x + threadIdx.x;
  if (i < n) out[i] = (_Float16)in[i];
}

// ---------------- input projection (K=10, trivial) ----------------
__global__ void k_input_proj(const float* __restrict__ x, const float* __restrict__ Win,
                             const float* __restrict__ bin, float* __restrict__ h) {
  int idx = blockIdx.x * blockDim.x + threadIdx.x;
  if (idx >= NNODES * HDIM) return;
  int n = idx >> 7, j = idx & (HDIM - 1);
  float s = bin[j];
#pragma unroll
  for (int i = 0; i < INDIM; ++i) s += x[n * INDIM + i] * Win[i * HDIM + j];
  h[idx] = s;
}

// ---------------- LayerNorm (f32 in -> f16 out) ----------------
__global__ void k_layernorm(const float* __restrict__ h, const float* __restrict__ g,
                            const float* __restrict__ b, _Float16* __restrict__ out) {
  int row = blockIdx.x;
  int t = threadIdx.x;  // 128 threads
  float x = h[(size_t)row * HDIM + t];
  float s = x, s2 = x * x;
#pragma unroll
  for (int m = 16; m >= 1; m >>= 1) {
    s  += __shfl_xor(s, m, 32);
    s2 += __shfl_xor(s2, m, 32);
  }
  __shared__ float red[2][4];
  if ((t & 31) == 0) { red[0][t >> 5] = s; red[1][t >> 5] = s2; }
  __syncthreads();
  float S = 0.f, S2 = 0.f;
#pragma unroll
  for (int i = 0; i < 4; ++i) { S += red[0][i]; S2 += red[1][i]; }
  float mu  = S * (1.f / HDIM);
  float var = S2 * (1.f / HDIM) - mu * mu;
  float y = (x - mu) * rsqrtf(var + 1e-5f) * g[t] + b[t];
  out[(size_t)row * HDIM + t] = (_Float16)y;
}

// ---------------- generic WMMA GEMM: one wave per 16x64 tile ----------------
// Register blocked: 1 A-frag reused by 4 B-frags -> 4 wmmas per K-step.
// mode 0: f16 out row-major   mode 1: f16 out transposed (Ch[col*M+row])
// mode 2: f32 residual add    (Cf[row*Nc+col] = res[...] + val)
__global__ __launch_bounds__(256)
void k_gemm(const _Float16* __restrict__ A, int lda,
            const _Float16* __restrict__ Bt, int ldb,
            const float* __restrict__ bias, const float* __restrict__ res,
            float* __restrict__ Cf, _Float16* __restrict__ Ch,
            int M, int Nc, int K, int mode, int relu) {
  int lane = threadIdx.x & 31;
  int gw = blockIdx.x * (blockDim.x >> 5) + (threadIdx.x >> 5);
  int ntn = Nc >> 6;                 // 64-wide n tiles
  int total = (M >> 4) * ntn;
  if (gw >= total) return;
  int m0 = (gw / ntn) << 4;
  int n0 = (gw - (gw / ntn) * ntn) << 6;
  v8f acc0 = vzero8(), acc1 = vzero8(), acc2 = vzero8(), acc3 = vzero8();
  for (int k0 = 0; k0 < K; k0 += 32) {
    v16h a  = load_a_frag(A, lda, m0, k0, lane);
    v16h b0 = load_b_frag(Bt, ldb, n0,      k0, lane);
    v16h b1 = load_b_frag(Bt, ldb, n0 + 16, k0, lane);
    v16h b2 = load_b_frag(Bt, ldb, n0 + 32, k0, lane);
    v16h b3 = load_b_frag(Bt, ldb, n0 + 48, k0, lane);
    acc0 = wmma_f16(a, b0, acc0);
    acc1 = wmma_f16(a, b1, acc1);
    acc2 = wmma_f16(a, b2, acc2);
    acc3 = wmma_f16(a, b3, acc3);
  }
  int col = lane & 15, hi = lane >> 4;
  v8f accs[4] = {acc0, acc1, acc2, acc3};
#pragma unroll
  for (int t = 0; t < 4; ++t) {
    int n = n0 + (t << 4) + col;
    float bb = bias[n];
#pragma unroll
    for (int j = 0; j < 8; ++j) {
      int row = m0 + j + (hi << 3);
      float v = accs[t][j] + bb;
      if (relu) v = fmaxf(v, 0.f);
      if (mode == 0)      Ch[(size_t)row * Nc + n] = (_Float16)v;
      else if (mode == 1) Ch[(size_t)n * M + row] = (_Float16)v;
      else                Cf[(size_t)row * Nc + n] =
                            res[(size_t)row * Nc + n] + v;
    }
  }
}

// ---------------- flash attention: one wave per (head, 16-query tile) ----------------
__global__ __launch_bounds__(256)
void k_attention(const _Float16* __restrict__ q16, const _Float16* __restrict__ k16,
                 const _Float16* __restrict__ vt16, _Float16* __restrict__ o16) {
  __shared__ _Float16 pbuf[8][16 * 32];
  int lane = threadIdx.x & 31, w = threadIdx.x >> 5;
  int gw = blockIdx.x * 8 + w;          // 0..1023
  int head = gw >> 8;                   // / (NNODES/16)
  int q0 = (gw & 255) << 4;
  int hoff = head * HEADD;
  int col = lane & 15, hi = lane >> 4;
  _Float16* pb = &pbuf[w][0];

  v16h qf = load_a_frag(q16, HDIM, q0, hoff, lane);
  const float scale = 0.17677669529663687f;  // 1/sqrt(32)
  float mrow[8], lrow[8];
#pragma unroll
  for (int j = 0; j < 8; ++j) { mrow[j] = -1e30f; lrow[j] = 0.f; }
  v8f acc0 = vzero8(), acc1 = vzero8();

  for (int kt = 0; kt < NNODES; kt += 32) {
    v16h kb0 = load_b_frag(k16, HDIM, kt,      hoff, lane);
    v16h kb1 = load_b_frag(k16, HDIM, kt + 16, hoff, lane);
    if (kt + 32 < NNODES)
      __builtin_prefetch(k16 + (size_t)(kt + 32 + col) * HDIM + hoff, 0, 1);
    v8f s0 = wmma_f16(qf, kb0, vzero8());
    v8f s1 = wmma_f16(qf, kb1, vzero8());
#pragma unroll
    for (int j = 0; j < 8; ++j) {
      float a = s0[j] * scale, b = s1[j] * scale;
      float t = fmaxf(a, b);
      t = fmaxf(t, __shfl_xor(t, 1, 32));
      t = fmaxf(t, __shfl_xor(t, 2, 32));
      t = fmaxf(t, __shfl_xor(t, 4, 32));
      t = fmaxf(t, __shfl_xor(t, 8, 32));
      float newm = fmaxf(mrow[j], t);
      float f  = __expf(mrow[j] - newm);
      float p0 = __expf(a - newm);
      float p1 = __expf(b - newm);
      float ts = p0 + p1;
      ts += __shfl_xor(ts, 1, 32);
      ts += __shfl_xor(ts, 2, 32);
      ts += __shfl_xor(ts, 4, 32);
      ts += __shfl_xor(ts, 8, 32);
      lrow[j] = lrow[j] * f + ts;
      mrow[j] = newm;
      acc0[j] *= f;
      acc1[j] *= f;
      int r = j + (hi << 3);  // C-layout row
      pb[r * 32 + col]      = (_Float16)p0;
      pb[r * 32 + 16 + col] = (_Float16)p1;
    }
    // same-wave LDS RAW: drain DS counter before re-reading P as an A-fragment
    asm volatile("s_wait_dscnt 0x0" ::: "memory");
    v16h pf  = load_a_frag(pb, 32, 0, 0, lane);
    v16h vb0 = load_b_frag(vt16, NNODES, hoff,      kt, lane);
    v16h vb1 = load_b_frag(vt16, NNODES, hoff + 16, kt, lane);
    acc0 = wmma_f16(pf, vb0, acc0);
    acc1 = wmma_f16(pf, vb1, acc1);
  }
#pragma unroll
  for (int j = 0; j < 8; ++j) {
    int row = q0 + j + (hi << 3);
    float inv = 1.f / lrow[j];
    o16[(size_t)row * HDIM + hoff + col]      = (_Float16)(acc0[j] * inv);
    o16[(size_t)row * HDIM + hoff + 16 + col] = (_Float16)(acc1[j] * inv);
  }
}

// ---------------- edge MLP: block = 16 edges, 8 waves = 8 col tiles ----------------
__global__ __launch_bounds__(256)
void k_edge_score(const _Float16* __restrict__ hemb, const int2* __restrict__ edges,
                  const _Float16* __restrict__ We1t, const float* __restrict__ be1,
                  const float* __restrict__ We2, const float* __restrict__ be2,
                  float* __restrict__ scores) {
  __shared__ float part[8][16];
  int lane = threadIdx.x & 31, w = threadIdx.x >> 5;
  int e0 = blockIdx.x << 4;
  int n0 = w << 4;
  int col = lane & 15, hi = lane >> 4;
  int2 ed = edges[e0 + col];  // A-fragment row for this lane
  v8f acc = vzero8();
#pragma unroll
  for (int k0 = 0; k0 < 2 * HDIM; k0 += 32) {
    int node = (k0 < HDIM) ? ed.x : ed.y;
    const _Float16* pa = hemb + (size_t)node * HDIM + (k0 & (HDIM - 1)) + (hi << 3);
    v8h a0 = *(const v8h*)pa;
    v8h a1 = *(const v8h*)(pa + 16);
    v16h a;
#pragma unroll
    for (int i = 0; i < 8; ++i) { a[i] = a0[i]; a[8 + i] = a1[i]; }
    v16h b = load_b_frag(We1t, 2 * HDIM, n0, k0, lane);
    acc = wmma_f16(a, b, acc);
  }
  float w2 = We2[n0 + col];
  float b1 = be1[n0 + col];
#pragma unroll
  for (int j = 0; j < 8; ++j) {
    float v = fmaxf(acc[j] + b1, 0.f) * w2;
    v += __shfl_xor(v, 1, 32);
    v += __shfl_xor(v, 2, 32);
    v += __shfl_xor(v, 4, 32);
    v += __shfl_xor(v, 8, 32);
    if (col == 0) part[w][j + (hi << 3)] = v;
  }
  __syncthreads();
  if (threadIdx.x < 16) {
    float s = be2[0];
#pragma unroll
    for (int q = 0; q < 8; ++q) s += part[q][threadIdx.x];
    scores[e0 + threadIdx.x] = s;
  }
}

// ---------------- global softmax over E ----------------
__global__ void k_pmax(const float* __restrict__ s, float* __restrict__ pmax, int n) {
  __shared__ float sh[8];
  float v = -1e30f;
  for (int i = blockIdx.x * blockDim.x + threadIdx.x; i < n; i += gridDim.x * blockDim.x)
    v = fmaxf(v, s[i]);
#pragma unroll
  for (int m = 16; m >= 1; m >>= 1) v = fmaxf(v, __shfl_xor(v, m, 32));
  if ((threadIdx.x & 31) == 0) sh[threadIdx.x >> 5] = v;
  __syncthreads();
  if (threadIdx.x == 0) {
    float r = sh[0];
    for (int i = 1; i < 8; ++i) r = fmaxf(r, sh[i]);
    pmax[blockIdx.x] = r;
  }
}

__global__ void k_rfinal(const float* __restrict__ in, float* __restrict__ gstat,
                         int n, int isSum) {
  __shared__ float sh[8];
  int t = threadIdx.x;
  float v = (t < n) ? in[t] : (isSum ? 0.f : -1e30f);
#pragma unroll
  for (int m = 16; m >= 1; m >>= 1) {
    float o = __shfl_xor(v, m, 32);
    v = isSum ? (v + o) : fmaxf(v, o);
  }
  if ((t & 31) == 0) sh[t >> 5] = v;
  __syncthreads();
  if (t == 0) {
    float r = sh[0];
    for (int i = 1; i < 8; ++i) r = isSum ? (r + sh[i]) : fmaxf(r, sh[i]);
    if (isSum) gstat[1] = 1.f / r; else gstat[0] = r;
  }
}

__global__ void k_exp(const float* __restrict__ s, const float* __restrict__ gstat,
                      float* __restrict__ out, float* __restrict__ psum, int n) {
  __shared__ float sh[8];
  float gmax = gstat[0];
  float acc = 0.f;
  for (int i = blockIdx.x * blockDim.x + threadIdx.x; i < n; i += gridDim.x * blockDim.x) {
    float e = __expf(s[i] - gmax);
    out[i] = e;
    acc += e;
  }
#pragma unroll
  for (int m = 16; m >= 1; m >>= 1) acc += __shfl_xor(acc, m, 32);
  if ((threadIdx.x & 31) == 0) sh[threadIdx.x >> 5] = acc;
  __syncthreads();
  if (threadIdx.x == 0) {
    float r = 0.f;
    for (int i = 0; i < 8; ++i) r += sh[i];
    psum[blockIdx.x] = r;
  }
}

__global__ void k_scale(float* __restrict__ out, const float* __restrict__ gstat, int n) {
  float inv = gstat[1];
  for (int i = blockIdx.x * blockDim.x + threadIdx.x; i < n; i += gridDim.x * blockDim.x)
    out[i] *= inv;
}

// ---------------- host orchestration ----------------
static void launch_gemm(hipStream_t stream, const _Float16* A, int lda,
                        const _Float16* Bt, int ldb, const float* bias,
                        const float* res, float* Cf, _Float16* Ch,
                        int M, int Nc, int K, int mode, int relu) {
  int tiles = (M >> 4) * (Nc >> 6);   // 16x64 tile per wave
  int blocks = (tiles + 7) >> 3;      // 8 waves (256 threads) per block
  k_gemm<<<blocks, 256, 0, stream>>>(A, lda, Bt, ldb, bias, res, Cf, Ch,
                                     M, Nc, K, mode, relu);
}

extern "C" void kernel_launch(void* const* d_in, const int* in_sizes, int n_in,
                              void* d_out, int out_size, void* d_ws, size_t ws_size,
                              hipStream_t stream) {
  (void)in_sizes; (void)n_in; (void)out_size; (void)ws_size;
  const float* x     = (const float*)d_in[0];
  const int2*  edges = (const int2*)d_in[1];
  const float* W_in  = (const float*)d_in[2];
  const float* b_in  = (const float*)d_in[3];
  const float* Wq = (const float*)d_in[4];  const float* bq = (const float*)d_in[5];
  const float* Wk = (const float*)d_in[6];  const float* bk = (const float*)d_in[7];
  const float* Wv = (const float*)d_in[8];  const float* bv = (const float*)d_in[9];
  const float* Wo = (const float*)d_in[10]; const float* bo = (const float*)d_in[11];
  const float* ln1_g = (const float*)d_in[12]; const float* ln1_b = (const float*)d_in[13];
  const float* ln2_g = (const float*)d_in[14]; const float* ln2_b = (const float*)d_in[15];
  const float* Wf1 = (const float*)d_in[16]; const float* bf1 = (const float*)d_in[17];
  const float* Wf2 = (const float*)d_in[18]; const float* bf2 = (const float*)d_in[19];
  const float* We1 = (const float*)d_in[20]; const float* be1 = (const float*)d_in[21];
  const float* We2 = (const float*)d_in[22]; const float* be2 = (const float*)d_in[23];
  float* probs = (float*)d_out;

  char* ws = (char*)d_ws;
  float*    h    = (float*)   (ws + 0);         // 4096*128 f32  (2 MB)
  _Float16* hn   = (_Float16*)(ws + 2097152);   // 4096*128 f16
  _Float16* q16  = (_Float16*)(ws + 3145728);
  _Float16* k16  = (_Float16*)(ws + 4194304);
  _Float16* vt16 = (_Float16*)(ws + 5242880);   // V transposed [128][4096]
  _Float16* o16  = (_Float16*)(ws + 6291456);
  _Float16* ff16 = (_Float16*)(ws + 7340032);   // 4096*512 f16 (4 MB)
  char*     wts  = ws + 11534336;               // per-layer f16 transposed weights
  _Float16* we1t = (_Float16*)(ws + 13107200);  // [128][256]
  float*    scores = (float*)(ws + 13172736);   // E f32
  float*    pmax = (float*)(ws + 13697024);
  float*    psum = (float*)(ws + 13698048);
  float*    gstat = (float*)(ws + 13699072);

  // ---- convert + transpose all weights to f16 (K-contiguous) ----
  for (int l = 0; l < NLAYER; ++l) {
    char* base = wts + (size_t)l * 393216;
    k_transpose_f16<<<64, 256, 0, stream>>>(Wq + l * 16384, (_Float16*)(base + 0),      128, 128);
    k_transpose_f16<<<64, 256, 0, stream>>>(Wk + l * 16384, (_Float16*)(base + 32768),  128, 128);
    k_transpose_f16<<<64, 256, 0, stream>>>(Wv + l * 16384, (_Float16*)(base + 65536),  128, 128);
    k_transpose_f16<<<64, 256, 0, stream>>>(Wo + l * 16384, (_Float16*)(base + 98304),  128, 128);
    k_transpose_f16<<<256, 256, 0, stream>>>(Wf1 + l * 65536, (_Float16*)(base + 131072), 128, 512);
    k_transpose_f16<<<256, 256, 0, stream>>>(Wf2 + l * 65536, (_Float16*)(base + 262144), 512, 128);
  }
  k_transpose_f16<<<128, 256, 0, stream>>>(We1, we1t, 256, 128);

  // ---- input projection ----
  k_input_proj<<<(NNODES * HDIM + 255) / 256, 256, 0, stream>>>(x, W_in, b_in, h);

  // ---- transformer layers ----
  for (int l = 0; l < NLAYER; ++l) {
    char* base = wts + (size_t)l * 393216;
    const _Float16* wqt  = (const _Float16*)(base + 0);
    const _Float16* wkt  = (const _Float16*)(base + 32768);
    const _Float16* wvt  = (const _Float16*)(base + 65536);
    const _Float16* wot  = (const _Float16*)(base + 98304);
    const _Float16* wf1t = (const _Float16*)(base + 131072);
    const _Float16* wf2t = (const _Float16*)(base + 262144);

    k_layernorm<<<NNODES, HDIM, 0, stream>>>(h, ln1_g + l * HDIM, ln1_b + l * HDIM, hn);
    launch_gemm(stream, hn, 128, wqt, 128, bq + l * HDIM, nullptr, nullptr, q16,
                NNODES, 128, 128, 0, 0);
    launch_gemm(stream, hn, 128, wkt, 128, bk + l * HDIM, nullptr, nullptr, k16,
                NNODES, 128, 128, 0, 0);
    launch_gemm(stream, hn, 128, wvt, 128, bv + l * HDIM, nullptr, nullptr, vt16,
                NNODES, 128, 128, 1, 0);  // transposed output for attention B-frags
    k_attention<<<(NHEADS * (NNODES / 16)) / 8, 256, 0, stream>>>(q16, k16, vt16, o16);
    launch_gemm(stream, o16, 128, wot, 128, bo + l * HDIM, h, h, nullptr,
                NNODES, 128, 128, 2, 0);  // residual add into h
    k_layernorm<<<NNODES, HDIM, 0, stream>>>(h, ln2_g + l * HDIM, ln2_b + l * HDIM, hn);
    launch_gemm(stream, hn, 128, wf1t, 128, bf1 + l * FFDIM, nullptr, nullptr, ff16,
                NNODES, 512, 128, 0, 1);  // ReLU
    launch_gemm(stream, ff16, 512, wf2t, 512, bf2 + l * HDIM, h, h, nullptr,
                NNODES, 128, 512, 2, 0);  // residual add into h
  }

  // ---- edge MLP + global softmax ----
  k_f32_to_f16<<<(NNODES * HDIM + 255) / 256, 256, 0, stream>>>(h, hn, NNODES * HDIM);
  k_edge_score<<<NEDGES / 16, 256, 0, stream>>>(hn, edges, we1t, be1, We2, be2, scores);
  k_pmax<<<256, 256, 0, stream>>>(scores, pmax, NEDGES);
  k_rfinal<<<1, 256, 0, stream>>>(pmax, gstat, 256, 0);
  k_exp<<<256, 256, 0, stream>>>(scores, gstat, probs, psum, NEDGES);
  k_rfinal<<<1, 256, 0, stream>>>(psum, gstat, 256, 1);
  k_scale<<<256, 256, 0, stream>>>(probs, gstat, NEDGES);
}
